// rgcn_layer_78829829751101
// MI455X (gfx1250) — matrix operand order repaired and checked by reading, not compiled, not run on hardware
//
#include <hip/hip_runtime.h>
#include <hip/hip_bf16.h>

// ---------------------------------------------------------------------------
// RGCN-ish bipartite layer for MI455X (gfx1250, wave32, WMMA + TDM).
// GEMMs use bf16x3 error-compensated v_wmma_f32_16x16x32_bf16 (hi*hi + hi*lo
// + lo*hi, fp32 accum ~= fp32 GEMM). Weights are split/swizzled ONCE into
// workspace in WMMA B-fragment order; each GEMM block stages them into LDS
// with a single Tensor Data Mover load (tensor_load_to_lds + s_wait_tensorcnt)
// when available, else a vectorized copy.
// ---------------------------------------------------------------------------

#define NUSER 50000
#define NITEM 50000
#define NTOT  100000
#define NEDGE 600000
#define DIM   128
#define HID_D 256
#define LN_EPS 1e-5f

typedef __attribute__((ext_vector_type(16))) __bf16       v16bf;
typedef __attribute__((ext_vector_type(8)))  float        v8f;
typedef __attribute__((ext_vector_type(4)))  float        v4f;
typedef __attribute__((ext_vector_type(4)))  unsigned int v4u;
typedef __attribute__((ext_vector_type(8)))  int          v8i;
typedef __attribute__((ext_vector_type(4)))  int          v4i;

#if __has_builtin(__builtin_amdgcn_tensor_load_to_lds) && \
    __has_builtin(__builtin_amdgcn_s_wait_tensorcnt)
#define USE_TDM 1
#endif

// ---------------------------------------------------------------------------
// One-shot: split fp32 weight (row-major Kd x Nd) into hi/lo bf16, swizzled
// into WMMA B-fragment order:  idx = (((kc*NT + nt)*32 + lane)*16 + j)
//   k = kc*32 + (lane>=16 ? 8 : 0) + (j<8 ? j : 8+j) ,  n = nt*16 + (lane&15)
// out[0 .. K*N)      = hi
// out[K*N .. 2*K*N)  = lo
// ---------------------------------------------------------------------------
__global__ __launch_bounds__(256) void split_weights(
    const float* __restrict__ W, __bf16* __restrict__ out, int Kd, int Nd)
{
    const int total = Kd * Nd;
    const int NT = Nd / 16;
    const int idx = blockIdx.x * 256 + threadIdx.x;
    if (idx >= total) return;
    const int j    = idx & 15;
    const int lane = (idx >> 4) & 31;
    const int nt   = (idx >> 9) % NT;
    const int kc   = (idx >> 9) / NT;
    const int k    = kc * 32 + ((lane & 16) ? 8 : 0) + (j < 8 ? j : 8 + j);
    const int n    = nt * 16 + (lane & 15);
    const float w  = W[(size_t)k * Nd + n];
    const __bf16 h = (__bf16)w;
    out[idx]         = h;
    out[total + idx] = (__bf16)(w - (float)h);
}

#ifdef USE_TDM
// 1D TDM DMA: copy `bytes` (multiple of 8, <= 512KB) from global to LDS.
// D# per CDNA5 ISA 8.3/8.4: group0 {count=1, lds_addr, global_addr, type=2},
// group1 {data_size=8B, tensor_dim0=tile_dim0=stride0=bytes/8, dim1=1}.
// Groups 2/3 (and the trailing group operand) are unused for <=2D tensors.
__device__ __forceinline__ void tdm_load_1d(const void* gsrc, void* ldsDst,
                                            unsigned bytes)
{
    const unsigned ne8 = bytes >> 3;  // 8-byte elements (tile_dim0 <= 65535)
    const unsigned long long ga = (unsigned long long)gsrc;
    const unsigned lds = (unsigned)(unsigned long long)ldsDst; // low32 = LDS off
    v4u g0;
    g0[0] = 1u;                                         // count=1, user mode
    g0[1] = lds;                                        // lds_addr
    g0[2] = (unsigned)(ga & 0xFFFFFFFFu);               // global_addr[31:0]
    g0[3] = (unsigned)((ga >> 32) & 0x01FFFFFFu)        // global_addr[56:32]
          | 0x80000000u;                                // type = 2 ("image")
    v8i g1;
    g1[0] = (int)(3u << 16);                            // data_size = 8B
    g1[1] = (int)((ne8 & 0xFFFFu) << 16);               // tensor_dim0[15:0]
    g1[2] = (int)((ne8 >> 16) | (1u << 16));            // tdim0[31:16], tdim1=1
    g1[3] = (int)(ne8 << 16);                           // tile_dim0
    g1[4] = 1;                                          // tile_dim1=1, tile_dim2=0
    g1[5] = (int)ne8;                                   // tensor_dim0_stride
    g1[6] = 0;
    g1[7] = 0;
    const v4i z4 = {0, 0, 0, 0};
    const v8i z8 = {0, 0, 0, 0, 0, 0, 0, 0};
    __builtin_amdgcn_tensor_load_to_lds(g0, g1, z4, z4, z8, 0);
}
#endif

// ---------------------------------------------------------------------------
// GEMM: C[M x N] = act(A[M x K] @ W[K x N] + bias), W pre-split/swizzled bf16.
// 256 threads = 8 wave32; each wave owns a 16-row strip across all N tiles.
// WMMA 16x16x32 bf16 layouts (CDNA5 ISA 7.12.2):
//   A: lanes 0-15 -> M=lane; elems j<8: K=khalf+j, j>=8: K=16+khalf+(j-8),
//      khalf=(lane>=16)*8
//   B: lane&15 = N; elem j: K same as A (khalf+j, j>=8: 16+khalf+(j-8))  (LDS pre-swizzled)
//   C/D: elem r -> M = r + (lane>=16)*8, N = lane&15
// ---------------------------------------------------------------------------
template <int K, int N, bool RELU>
__global__ __launch_bounds__(256) void gemm_bf16x3(
    const float* __restrict__ A, const __bf16* __restrict__ Wsw,
    const float* __restrict__ bias, float* __restrict__ C, int M)
{
    constexpr int KC = K / 32;
    constexpr int NT = N / 16;
    constexpr int WBYTES = 2 * K * N * 2;     // hi+lo bf16 image

    extern __shared__ __bf16 lds[];           // [2][KC][NT][32][16] bf16
    __bf16* whi = lds;
    __bf16* wlo = lds + K * N;

    const int tid = threadIdx.x;

    // ---- Stage pre-swizzled weights into LDS --------------------------------
#ifdef USE_TDM
    if (tid < 32) {                            // wave 0 issues one TDM DMA
        tdm_load_1d(Wsw, lds, WBYTES);
        __builtin_amdgcn_s_wait_tensorcnt(0);
    }
#else
    {
        const v4u* g = (const v4u*)Wsw;
        v4u* l = (v4u*)lds;
        for (int i = tid; i < WBYTES / 16; i += 256) l[i] = g[i];
    }
#endif
    __syncthreads();

    const int lane     = tid & 31;
    const int wave     = tid >> 5;
    const int rowStrip = blockIdx.x * 128 + wave * 16;
    if (rowStrip >= M) return;                 // uniform per-wave exit

    const int row  = rowStrip + (lane & 15);
    const int rowA = row < M ? row : M - 1;    // clamp reads; stores guarded
    const float* aptr = A + (size_t)rowA * K;
    const int khalf = (lane >> 4) * 8;

    // Preload + split all A fragments (resident in VGPRs).
    v16bf ah[KC], al[KC];
#pragma unroll
    for (int kc = 0; kc < KC; ++kc) {
        const float* p = aptr + kc * 32 + khalf;
        v4f x0 = *(const v4f*)(p);
        v4f x1 = *(const v4f*)(p + 4);
        v4f x2 = *(const v4f*)(p + 16);
        v4f x3 = *(const v4f*)(p + 20);
        float xs[16];
#pragma unroll
        for (int j = 0; j < 4; ++j) {
            xs[j] = x0[j]; xs[4 + j] = x1[j]; xs[8 + j] = x2[j]; xs[12 + j] = x3[j];
        }
#pragma unroll
        for (int j = 0; j < 16; ++j) {
            const float  x = xs[j];
            const __bf16 h = (__bf16)x;
            ah[kc][j] = h;
            al[kc][j] = (__bf16)(x - (float)h);
        }
    }

    const int nbase = lane & 15;
    const int mtop  = (lane >> 4) * 8;

    for (int nt = 0; nt < NT; ++nt) {
        const float binit = bias ? bias[nt * 16 + nbase] : 0.0f;
        v8f acc, accc;                          // main + correction chains
#pragma unroll
        for (int r = 0; r < 8; ++r) { acc[r] = binit; accc[r] = 0.0f; }

#pragma unroll
        for (int kc = 0; kc < KC; ++kc) {
            const v16bf bh = *(const v16bf*)(whi + ((size_t)(kc * NT + nt) * 32 + lane) * 16);
            const v16bf bl = *(const v16bf*)(wlo + ((size_t)(kc * NT + nt) * 32 + lane) * 16);
            accc = __builtin_amdgcn_wmma_f32_16x16x32_bf16(
                false, ah[kc], false, bl, (short)0, accc, false, false);
            accc = __builtin_amdgcn_wmma_f32_16x16x32_bf16(
                false, al[kc], false, bh, (short)0, accc, false, false);
            acc  = __builtin_amdgcn_wmma_f32_16x16x32_bf16(
                false, ah[kc], false, bh, (short)0, acc, false, false);
        }

#pragma unroll
        for (int r = 0; r < 8; ++r) {
            const int rr = rowStrip + mtop + r;
            if (rr < M) {
                float v = acc[r] + accc[r];
                if (RELU) v = fmaxf(v, 0.0f);
                C[(size_t)rr * N + nt * 16 + nbase] = v;
            }
        }
    }
}

// ---------------------------------------------------------------------------
// msg init: rows [0,NUSER) = b_i2u broadcast, rows [NUSER,NTOT) = b_u2i.
// ---------------------------------------------------------------------------
__global__ __launch_bounds__(256) void init_msg(
    float* __restrict__ MSG, const float* __restrict__ b_i2u,
    const float* __restrict__ b_u2i)
{
    const long long idx = (long long)blockIdx.x * 256 + threadIdx.x;
    if (idx >= (long long)NTOT * DIM) return;
    const int c   = (int)(idx & (DIM - 1));
    const int row = (int)(idx >> 7);
    MSG[idx] = (row < NUSER) ? b_i2u[c] : b_u2i[c];
}

// ---------------------------------------------------------------------------
// Edge scatter-add: one wave per edge, float4 gather + 4 native f32 atomics.
// ---------------------------------------------------------------------------
__global__ __launch_bounds__(256) void scatter_edges(
    const float* __restrict__ T, const int* __restrict__ src,
    const int* __restrict__ dst, float* __restrict__ out, int ne)
{
    const int e = blockIdx.x * 8 + (threadIdx.x >> 5);
    if (e >= ne) return;
    const int lane = threadIdx.x & 31;
    const int s = src[e];
    const int d = dst[e];
    const v4f v = *(const v4f*)(T + (size_t)s * DIM + lane * 4);
    float* p = out + (size_t)d * DIM + lane * 4;
    unsafeAtomicAdd(p + 0, v[0]);
    unsafeAtomicAdd(p + 1, v[1]);
    unsafeAtomicAdd(p + 2, v[2]);
    unsafeAtomicAdd(p + 3, v[3]);
}

// ---------------------------------------------------------------------------
// h = S + msg; LayerNorm(D=128) -> ReLU -> + feat residual; in place to MSG.
// One wave32 per row, 4 values/lane, shfl_xor butterfly reductions.
// ---------------------------------------------------------------------------
__global__ __launch_bounds__(256) void combine_ln(
    const float* __restrict__ S, float* __restrict__ MSG,
    const float* __restrict__ feat_user, const float* __restrict__ feat_item,
    const float* __restrict__ gu, const float* __restrict__ bu,
    const float* __restrict__ gi, const float* __restrict__ bi, int nrows)
{
    const int row = blockIdx.x * 8 + (threadIdx.x >> 5);
    if (row >= nrows) return;
    const int lane = threadIdx.x & 31;
    const bool isU = row < NUSER;
    const float* feat = isU ? (feat_user + (size_t)row * DIM)
                            : (feat_item + (size_t)(row - NUSER) * DIM);
    const float* g = isU ? gu : gi;
    const float* b = isU ? bu : bi;
    const float* srow = S + (size_t)row * DIM;
    float* mrow = MSG + (size_t)row * DIM;

    const int c0 = lane * 4;
    const v4f sv = *(const v4f*)(srow + c0);
    const v4f mv = *(const v4f*)(mrow + c0);
    float x[4];
    float sum = 0.0f;
#pragma unroll
    for (int j = 0; j < 4; ++j) { x[j] = sv[j] + mv[j]; sum += x[j]; }
#pragma unroll
    for (int off = 16; off > 0; off >>= 1) sum += __shfl_xor(sum, off, 32);
    const float mean = sum * (1.0f / 128.0f);

    float var = 0.0f;
#pragma unroll
    for (int j = 0; j < 4; ++j) { const float d = x[j] - mean; var += d * d; }
#pragma unroll
    for (int off = 16; off > 0; off >>= 1) var += __shfl_xor(var, off, 32);
    const float rs = rsqrtf(var * (1.0f / 128.0f) + LN_EPS);

    v4f o;
#pragma unroll
    for (int j = 0; j < 4; ++j) {
        const int c = c0 + j;
        float y = (x[j] - mean) * rs * g[c] + b[c];
        o[j] = fmaxf(y, 0.0f) + feat[c];
    }
    *(v4f*)(mrow + c0) = o;
}

// ---------------------------------------------------------------------------
// Host orchestration.
// ws layout: TS[NTOT*128]f | MSG[NTOT*128]f | HID[NTOT*256]f | Wsplit[512KB]
// ---------------------------------------------------------------------------
extern "C" void kernel_launch(void* const* d_in, const int* in_sizes, int n_in,
                              void* d_out, int out_size, void* d_ws, size_t ws_size,
                              hipStream_t stream)
{
    const float* feat_user   = (const float*)d_in[0];
    const float* feat_item   = (const float*)d_in[1];
    const float* W_u2i       = (const float*)d_in[2];
    const float* b_u2i       = (const float*)d_in[3];
    const float* W_i2u       = (const float*)d_in[4];
    const float* b_i2u       = (const float*)d_in[5];
    const float* self_w_user = (const float*)d_in[6];
    const float* self_w_item = (const float*)d_in[7];
    const float* ln_g_user   = (const float*)d_in[8];
    const float* ln_b_user   = (const float*)d_in[9];
    const float* ln_g_item   = (const float*)d_in[10];
    const float* ln_b_item   = (const float*)d_in[11];
    const float* ffn_w1      = (const float*)d_in[12];
    const float* ffn_b1      = (const float*)d_in[13];
    const float* ffn_w2      = (const float*)d_in[14];
    const float* ffn_b2      = (const float*)d_in[15];
    const int*   src_u2i     = (const int*)d_in[16];
    const int*   dst_u2i     = (const int*)d_in[17];
    const int*   src_i2u     = (const int*)d_in[18];
    const int*   dst_i2u     = (const int*)d_in[19];

    float* TS  = (float*)d_ws;                   // transformed feats, later self
    float* MSG = TS  + (size_t)NTOT * DIM;       // msg accum -> h_ln
    float* HID = MSG + (size_t)NTOT * DIM;       // FFN hidden

    // Pre-split/swizzled bf16 weight images (hi block then lo block each).
    __bf16* WSP   = (__bf16*)(HID + (size_t)NTOT * HID_D);
    __bf16* sw_u2i  = WSP;                       // 2*128*128
    __bf16* sw_i2u  = sw_u2i + 2 * DIM * DIM;
    __bf16* sw_su   = sw_i2u + 2 * DIM * DIM;
    __bf16* sw_si   = sw_su  + 2 * DIM * DIM;
    __bf16* sw_w1   = sw_si  + 2 * DIM * DIM;    // 2*128*256
    __bf16* sw_w2   = sw_w1  + 2 * DIM * HID_D;  // 2*256*128

    (void)hipFuncSetAttribute(reinterpret_cast<const void*>(gemm_bf16x3<128, 128, false>),
                              hipFuncAttributeMaxDynamicSharedMemorySize, 64 * 1024);
    (void)hipFuncSetAttribute(reinterpret_cast<const void*>(gemm_bf16x3<128, 256, true>),
                              hipFuncAttributeMaxDynamicSharedMemorySize, 128 * 1024);
    (void)hipFuncSetAttribute(reinterpret_cast<const void*>(gemm_bf16x3<256, 128, false>),
                              hipFuncAttributeMaxDynamicSharedMemorySize, 128 * 1024);

    const dim3 blk(256);

    // 0) One-shot weight split/swizzle (~512 KB total, trivial).
    split_weights<<<(DIM * DIM + 255) / 256, blk, 0, stream>>>(W_u2i, sw_u2i, DIM, DIM);
    split_weights<<<(DIM * DIM + 255) / 256, blk, 0, stream>>>(W_i2u, sw_i2u, DIM, DIM);
    split_weights<<<(DIM * DIM + 255) / 256, blk, 0, stream>>>(self_w_user, sw_su, DIM, DIM);
    split_weights<<<(DIM * DIM + 255) / 256, blk, 0, stream>>>(self_w_item, sw_si, DIM, DIM);
    split_weights<<<(DIM * HID_D + 255) / 256, blk, 0, stream>>>(ffn_w1, sw_w1, DIM, HID_D);
    split_weights<<<(HID_D * DIM + 255) / 256, blk, 0, stream>>>(ffn_w2, sw_w2, HID_D, DIM);

    // 1) Message transforms: T_u = feat_user @ W_u2i ; T_i = feat_item @ W_i2u
    gemm_bf16x3<128, 128, false><<<(NUSER + 127) / 128, blk, 64 * 1024, stream>>>(
        feat_user, sw_u2i, nullptr, TS, NUSER);
    gemm_bf16x3<128, 128, false><<<(NITEM + 127) / 128, blk, 64 * 1024, stream>>>(
        feat_item, sw_i2u, nullptr, TS + (size_t)NUSER * DIM, NITEM);

    // 2) Init msg accumulators with biases (msg = segsum + b).
    init_msg<<<((long long)NTOT * DIM + 255) / 256, blk, 0, stream>>>(MSG, b_i2u, b_u2i);

    // 3) Edge scatter-adds (L2-resident atomics).
    scatter_edges<<<(NEDGE + 7) / 8, blk, 0, stream>>>(
        TS, src_u2i, dst_u2i, MSG + (size_t)NUSER * DIM, NEDGE);   // user->item
    scatter_edges<<<(NEDGE + 7) / 8, blk, 0, stream>>>(
        TS + (size_t)NUSER * DIM, src_i2u, dst_i2u, MSG, NEDGE);   // item->user

    // 4) Self-loop transforms, reusing TS (scatters already consumed it).
    gemm_bf16x3<128, 128, false><<<(NUSER + 127) / 128, blk, 64 * 1024, stream>>>(
        feat_user, sw_su, nullptr, TS, NUSER);
    gemm_bf16x3<128, 128, false><<<(NITEM + 127) / 128, blk, 64 * 1024, stream>>>(
        feat_item, sw_si, nullptr, TS + (size_t)NUSER * DIM, NITEM);

    // 5) h = S + msg -> LN -> ReLU -> + feat   (in place into MSG)
    combine_ln<<<(NTOT + 7) / 8, blk, 0, stream>>>(
        TS, MSG, feat_user, feat_item,
        ln_g_user, ln_b_user, ln_g_item, ln_b_item, NTOT);

    // 6) FFN: HID = relu(MSG @ W1 + b1) ; out = HID @ W2 + b2
    gemm_bf16x3<128, 256, true><<<(NTOT + 127) / 128, blk, 128 * 1024, stream>>>(
        MSG, sw_w1, ffn_b1, HID, NTOT);
    gemm_bf16x3<256, 128, false><<<(NTOT + 127) / 128, blk, 128 * 1024, stream>>>(
        HID, sw_w2, ffn_b2, (float*)d_out, NTOT);
}
